// BiAttentionEncoder_30133490549690
// MI455X (gfx1250) — matrix operand order repaired and checked by reading, not compiled, not run on hardware
//
#include <hip/hip_runtime.h>

// ---------- types ----------
typedef _Float16 h16;
typedef __attribute__((ext_vector_type(16))) h16   v16h;
typedef __attribute__((ext_vector_type(8)))  float v8f;
typedef __attribute__((ext_vector_type(4)))  unsigned int u32x4;

#define DI __device__ __forceinline__

// ---------- model dims ----------
#define VV   30000
#define EE   300
#define EP   320
#define HH   512
#define H2   1024
#define H3   1536
#define H4   2048
#define BB   64
#define STT  128
#define SNN  512
#define MT   (BB*STT)     // 8192
#define MN   (BB*SNN)     // 32768
#define OUTROW (STT+SNN)  // 640

// ---------- WMMA fragment loaders (CDNA5 16x16x32 f16 layouts) ----------
// A: 16(M)x32(K). lanes 0-15: M=lane, K blocks {0..7,16..23}; lanes 16-31: K {8..15,24..31}
DI v16h load_a_frag(const h16* base, int ldm, int lane) {
  int m = lane & 15, hf = lane >> 4;
  union { v16h v; u32x4 q[2]; } u;
  u.q[0] = *(const u32x4*)(base + (long)m * ldm + hf * 8);
  u.q[1] = *(const u32x4*)(base + (long)m * ldm + 16 + hf * 8);
  return u.v;
}
// B: 32(K)x16(N), K-major storage. lanes 0-15: N=lane, K {0..7,16..23}; lanes 16-31: K {8..15,24..31}
DI v16h load_b_frag(const h16* base, int ldn, int lane) {
  int n = lane & 15, hf = lane >> 4;
  v16h v;
#pragma unroll
  for (int s = 0; s < 16; ++s) {
    int k = (s < 8) ? (hf * 8 + s) : (16 + hf * 8 + (s - 8));
    v[s] = base[(long)k * ldn + n];
  }
  return v;
}
DI float sigm(float x) { return 1.0f / (1.0f + __expf(-x)); }

// ---------- generic WMMA GEMM:  C[M,N] = A16[M,K](lda) * Bt16[K,N](ldb) + bias ----------
constexpr int EP_F32 = 0, EP_F16 = 1, EP_F16T = 2, EP_TANH_F32 = 3;

template<int MODE>
__global__ __launch_bounds__(256) void gemm_wmma(
    const h16* __restrict__ A,  long lda, long bsA,
    const h16* __restrict__ Bt, long ldb, long bsB,
    const float* __restrict__ bias,
    void* __restrict__ Out, long bsO,
    int M, int N, int K,
    long RB, long OBS, long ORS)
{
  __shared__ __align__(16) h16 As[64 * 32];
  __shared__ __align__(16) h16 Bs[32 * 64];
  const int z = blockIdx.z;
  A  += (long)z * bsA;
  Bt += (long)z * bsB;
  const int m0 = blockIdx.x * 64, n0 = blockIdx.y * 64;
  const int tid = threadIdx.x, lane = tid & 31, w = tid >> 5;
  const int wm = w & 3, wn = w >> 2;           // 4x2 wave grid over 64x64 tile
  v8f c0 = {}, c1 = {};

  for (int kb = 0; kb < K; kb += 32) {
    { // stage A tile 64x32
      int r = tid >> 2, c8 = (tid & 3) * 8;
      *(u32x4*)&As[r * 32 + c8] = *(const u32x4*)&A[(long)(m0 + r) * lda + kb + c8];
      if (kb + 32 < K) __builtin_prefetch(&A[(long)(m0 + r) * lda + kb + 32 + c8], 0, 0);
    }
    { // stage B tile 32x64 (K-major source)
      int r = tid >> 3, c8 = (tid & 7) * 8;
      *(u32x4*)&Bs[r * 64 + c8] = *(const u32x4*)&Bt[(long)(kb + r) * ldb + n0 + c8];
      if (kb + 32 < K) __builtin_prefetch(&Bt[(long)(kb + 32 + r) * ldb + n0 + c8], 0, 0);
    }
    __syncthreads();
    v16h a  = load_a_frag(&As[wm * 16 * 32], 32, lane);
    v16h b0 = load_b_frag(&Bs[wn * 32],      64, lane);
    v16h b1 = load_b_frag(&Bs[wn * 32 + 16], 64, lane);
    c0 = __builtin_amdgcn_wmma_f32_16x16x32_f16(false, a, false, b0, (short)0, c0, false, false);
    c1 = __builtin_amdgcn_wmma_f32_16x16x32_f16(false, a, false, b1, (short)0, c1, false, false);
    __syncthreads();
  }

  const int hf = lane >> 4, nl = lane & 15;
#pragma unroll
  for (int p = 0; p < 2; ++p) {
    v8f c = p ? c1 : c0;
    int nn = n0 + wn * 32 + p * 16 + nl;
    float bv = bias ? bias[nn] : 0.0f;
#pragma unroll
    for (int v = 0; v < 8; ++v) {
      int m = m0 + wm * 16 + v + 8 * hf;
      float val = c[v] + bv;
      long off;
      if (MODE == EP_F16T) off = ((long)m / RB) * OBS + (long)nn * ORS + ((long)m % RB);
      else                 off = ((long)m / RB) * OBS + ((long)m % RB) * ORS + nn;
      off += (long)z * bsO;
      if      (MODE == EP_F32)      ((float*)Out)[off] = val;
      else if (MODE == EP_TANH_F32) ((float*)Out)[off] = tanhf(val);
      else                          ((h16*)Out)[off]   = (h16)val;
    }
  }
}

// ---------- weight cast/transpose: Wt16[Kp][N] = W[N][K]^T, zero-pad K->Kp ----------
__global__ void castT_kernel(const float* __restrict__ W, h16* __restrict__ Wt,
                             int N, int K, int Kp) {
  long idx = (long)blockIdx.x * blockDim.x + threadIdx.x;
  if (idx >= (long)Kp * N) return;
  int n = (int)(idx % N), k = (int)(idx / N);
  Wt[idx] = (k < K) ? (h16)W[(long)n * K + k] : (h16)0.0f;
}

// ---------- embedding gather -> padded f16 rows ----------
__global__ void embed_kernel(const int* __restrict__ toks, const float* __restrict__ embW,
                             h16* __restrict__ out, long nrows) {
  long idx = (long)blockIdx.x * blockDim.x + threadIdx.x;
  if (idx >= nrows * EP) return;
  long r = idx / EP; int c = (int)(idx % EP);
  int tk = toks[r];
  tk = (tk > VV) ? 3 : tk;
  out[idx] = (c < EE) ? (h16)embW[(long)tk * EE + c] : (h16)0.0f;
}

// ---------- GRU scan: 1 workgroup (16 waves, 512 thr) per (gru,dir) ----------
// h kept f32 in per-wave registers + f16 LDS mirror for the WMMA A operand.
// One J-tile at a time -> only 3 live zero-initialized accumulators; bhh biases
// are added as scalars in the gate epilogue (no bias-splat vectors to spill).
struct ScanArgs {
  const h16*   xp[4];     // [B*T][1536] f16
  const h16*   whhT[4];   // [512][1536] K-major f16
  const float* bhh[4];    // [1536]
  h16*         cat[4];    // [B*T][2048], write cols coloff..coloff+511
  float*       hid[4];    // [B][512]
  const int*   len[4];
  int T[4]; int rev[4]; int coloff[4];
};

__global__ __launch_bounds__(512) void gru_scan(ScanArgs args) {
  const int s = blockIdx.x;
  const h16*   xp    = args.xp[s];
  const h16*   whhT  = args.whhT[s];
  const float* bhh   = args.bhh[s];
  h16*         cat   = args.cat[s];
  float*       hid   = args.hid[s];
  const int*   lenp  = args.len[s];
  const int T = args.T[s], rev = args.rev[s], coloff = args.coloff[s];

  __shared__ __align__(16) h16 hsh[64 * 512];   // 64KB f16 mirror of h
  const int tid = threadIdx.x, lane = tid & 31, w = tid >> 5;   // 16 waves
  const int mt = w & 3, jg = w >> 2;            // wave: m-tile mt, j-tiles jg*8 .. jg*8+7
  const int hf = lane >> 4, nl = lane & 15;

  u32x4 zz = {0u, 0u, 0u, 0u};
  for (int i = tid; i < 64 * 512 / 8; i += 512) ((u32x4*)hsh)[i] = zz;

  int blocal[8]; int mylen[8];
#pragma unroll
  for (int v = 0; v < 8; ++v) { blocal[v] = mt * 16 + v + 8 * hf; mylen[v] = lenp[blocal[v]]; }

  float bh[8][3];
#pragma unroll
  for (int i = 0; i < 8; ++i) {
    int col = (jg * 8 + i) * 16 + nl;
#pragma unroll
    for (int g = 0; g < 3; ++g) bh[i][g] = bhh[g * 512 + col];
  }
  v8f h[8] = {};
  __syncthreads();

  for (int step = 0; step < T; ++step) {
    const int t = rev ? (T - 1 - step) : step;

#pragma unroll
    for (int i = 0; i < 8; ++i) {          // one J-tile at a time
      const int jt = jg * 8 + i;
      v8f a0 = {}, a1 = {}, a2 = {};       // zero-init accumulators (no bias splat)

      for (int kb = 0; kb < 512; kb += 32) {
        v16h a  = load_a_frag(&hsh[(mt * 16) * 512 + kb], 512, lane);
        const h16* wb = &whhT[(long)kb * H3 + jt * 16];
        v16h b0 = load_b_frag(wb,            H3, lane);
        v16h b1 = load_b_frag(wb + 512,      H3, lane);
        v16h b2 = load_b_frag(wb + 1024,     H3, lane);
        a0 = __builtin_amdgcn_wmma_f32_16x16x32_f16(false, a, false, b0, (short)0, a0, false, false);
        a1 = __builtin_amdgcn_wmma_f32_16x16x32_f16(false, a, false, b1, (short)0, a1, false, false);
        a2 = __builtin_amdgcn_wmma_f32_16x16x32_f16(false, a, false, b2, (short)0, a2, false, false);
      }

      const int col = jt * 16 + nl;
      const float br = bh[i][0], bz = bh[i][1], bn = bh[i][2];
#pragma unroll
      for (int v = 0; v < 8; ++v) {
        int b = blocal[v];
        long row = (long)b * T + t;
        const h16* x = &xp[row * H3];
        float xr = (float)x[col], xz = (float)x[512 + col], xn = (float)x[1024 + col];
        float r = sigm(xr + a0[v] + br);
        float z = sigm(xz + a1[v] + bz);
        float n = tanhf(xn + r * (a2[v] + bn));
        float hn = (1.0f - z) * n + z * h[i][v];
        bool valid = (t < mylen[v]);
        float hv = valid ? hn : h[i][v];
        h[i][v] = hv;                                   // register update only
        cat[row * H4 + coloff + col] = (h16)(valid ? hv : 0.0f);
      }
    }

    __syncthreads();   // everyone done READING hsh (old h)
#pragma unroll
    for (int i = 0; i < 8; ++i) {
      const int col = (jg * 8 + i) * 16 + nl;
#pragma unroll
      for (int v = 0; v < 8; ++v) hsh[blocal[v] * 512 + col] = (h16)h[i][v];
    }
    __syncthreads();   // hsh fully updated for next step
  }

#pragma unroll
  for (int i = 0; i < 8; ++i) {
    const int col = (jg * 8 + i) * 16 + nl;
#pragma unroll
    for (int v = 0; v < 8; ++v) hid[blocal[v] * 512 + col] = h[i][v];
  }
}

// ---------- row softmax f32 -> f16 probs ----------
__global__ __launch_bounds__(256) void softmax_rows(const float* __restrict__ S,
                                                    h16* __restrict__ P, int ncols) {
  long row = blockIdx.x;
  const float* src = S + row * ncols;
  h16* dst = P + row * ncols;
  __shared__ float red[256];
  int tid = threadIdx.x;
  float mx = -1e30f;
  for (int c = tid; c < ncols; c += 256) mx = fmaxf(mx, src[c]);
  red[tid] = mx; __syncthreads();
  for (int o = 128; o > 0; o >>= 1) { if (tid < o) red[tid] = fmaxf(red[tid], red[tid + o]); __syncthreads(); }
  mx = red[0]; __syncthreads();
  float sum = 0.0f;
  for (int c = tid; c < ncols; c += 256) sum += __expf(src[c] - mx);
  red[tid] = sum; __syncthreads();
  for (int o = 128; o > 0; o >>= 1) { if (tid < o) red[tid] += red[tid + o]; __syncthreads(); }
  float inv = 1.0f / red[0];
  for (int c = tid; c < ncols; c += 256) dst[c] = (h16)(__expf(src[c] - mx) * inv);
}

// ---------- comb linear: hidden = cat([hid_t,hid_n],-1) @ comb_W.T + b  -> [2,B,512] ----------
__global__ void comb_kernel(const float* __restrict__ hid_t, const float* __restrict__ hid_n,
                            const float* __restrict__ W, const float* __restrict__ bias,
                            float* __restrict__ out) {
  int idx = blockIdx.x * blockDim.x + threadIdx.x;
  if (idx >= 2 * BB * HH) return;
  int j = idx & 511, b = (idx >> 9) & 63, d = idx >> 15;
  const float* ht = hid_t + ((long)d * BB + b) * HH;
  const float* hn = hid_n + ((long)d * BB + b) * HH;
  float s = bias[j];
  const float* wr = W + (long)j * H2;
  for (int k = 0; k < HH; ++k) s += ht[k] * wr[k];
  for (int k = 0; k < HH; ++k) s += hn[k] * wr[HH + k];
  out[idx] = s;
}

// ---------- host ----------
extern "C" void kernel_launch(void* const* d_in, const int* in_sizes, int n_in,
                              void* d_out, int out_size, void* d_ws, size_t ws_size,
                              hipStream_t stream) {
  (void)in_sizes; (void)n_in; (void)out_size; (void)ws_size;
  const float* emb_W   = (const float*)d_in[0];
  const float* Wih[4]  = { (const float*)d_in[1], (const float*)d_in[5],
                           (const float*)d_in[9], (const float*)d_in[13] };
  const float* Whh[4]  = { (const float*)d_in[2], (const float*)d_in[6],
                           (const float*)d_in[10], (const float*)d_in[14] };
  const float* bih[4]  = { (const float*)d_in[3], (const float*)d_in[7],
                           (const float*)d_in[11], (const float*)d_in[15] };
  const float* bhh[4]  = { (const float*)d_in[4], (const float*)d_in[8],
                           (const float*)d_in[12], (const float*)d_in[16] };
  const float* biatt_W = (const float*)d_in[17]; const float* biatt_b = (const float*)d_in[18];
  const float* wt_W = (const float*)d_in[19];    const float* wt_b = (const float*)d_in[20];
  const float* wn_W = (const float*)d_in[21];    const float* wn_b = (const float*)d_in[22];
  const float* comb_W = (const float*)d_in[23];  const float* comb_b = (const float*)d_in[24];
  const int* tok_t = (const int*)d_in[25]; const int* tok_n = (const int*)d_in[26];
  const int* len_t = (const int*)d_in[27]; const int* len_n = (const int*)d_in[28];

  // workspace carve-up
  char* ws = (char*)d_ws;
  size_t off = 0;
  auto alloc = [&](size_t bytes) { off = (off + 255) & ~(size_t)255; size_t o = off; off += bytes; return o; };
  h16* wihT[4]; for (int i = 0; i < 4; ++i) wihT[i] = (h16*)(ws + alloc((size_t)EP * H3 * 2));
  h16* whhT[4]; for (int i = 0; i < 4; ++i) whhT[i] = (h16*)(ws + alloc((size_t)HH * H3 * 2));
  h16* biattT = (h16*)(ws + alloc((size_t)H2 * H2 * 2));
  h16* wtT    = (h16*)(ws + alloc((size_t)H4 * H2 * 2));
  h16* wnT    = (h16*)(ws + alloc((size_t)H4 * H2 * 2));
  h16* embT   = (h16*)(ws + alloc((size_t)MT * EP * 2));
  h16* embN   = (h16*)(ws + alloc((size_t)MN * EP * 2));
  h16* xp[4];
  xp[0] = (h16*)(ws + alloc((size_t)MT * H3 * 2));
  xp[1] = (h16*)(ws + alloc((size_t)MT * H3 * 2));
  xp[2] = (h16*)(ws + alloc((size_t)MN * H3 * 2));
  xp[3] = (h16*)(ws + alloc((size_t)MN * H3 * 2));
  h16* catT = (h16*)(ws + alloc((size_t)MT * H4 * 2));
  h16* catN = (h16*)(ws + alloc((size_t)MN * H4 * 2));
  float* hidT = (float*)(ws + alloc((size_t)2 * BB * HH * 4));
  float* hidN = (float*)(ws + alloc((size_t)2 * BB * HH * 4));
  h16* tnT  = (h16*)(ws + alloc((size_t)BB * H2 * SNN * 2));
  h16* ttT  = (h16*)(ws + alloc((size_t)BB * H2 * STT * 2));
  float* scTN = (float*)(ws + alloc((size_t)BB * STT * SNN * 4));
  float* scNT = (float*)(ws + alloc((size_t)BB * SNN * STT * 4));
  h16* pTN = (h16*)(ws + alloc((size_t)BB * STT * SNN * 2));
  h16* pNT = (h16*)(ws + alloc((size_t)BB * SNN * STT * 2));

  auto cgrid = [](long n) { return dim3((unsigned)((n + 255) / 256)); };

  // 1) weights -> f16 K-major
  for (int i = 0; i < 4; ++i)
    castT_kernel<<<cgrid((long)EP * H3), 256, 0, stream>>>(Wih[i], wihT[i], H3, EE, EP);
  for (int i = 0; i < 4; ++i)
    castT_kernel<<<cgrid((long)HH * H3), 256, 0, stream>>>(Whh[i], whhT[i], H3, HH, HH);
  castT_kernel<<<cgrid((long)H2 * H2), 256, 0, stream>>>(biatt_W, biattT, H2, H2, H2);
  castT_kernel<<<cgrid((long)H4 * H2), 256, 0, stream>>>(wt_W, wtT, H2, H4, H4);
  castT_kernel<<<cgrid((long)H4 * H2), 256, 0, stream>>>(wn_W, wnT, H2, H4, H4);

  // 2) embeddings
  embed_kernel<<<cgrid((long)MT * EP), 256, 0, stream>>>(tok_t, emb_W, embT, MT);
  embed_kernel<<<cgrid((long)MN * EP), 256, 0, stream>>>(tok_n, emb_W, embN, MN);

  // 3) input projections xp = emb @ Wih^T + bih   (f16 out, row stride 1536)
  for (int i = 0; i < 4; ++i) {
    const h16* E = (i < 2) ? embT : embN;
    int M = (i < 2) ? MT : MN;
    gemm_wmma<EP_F16><<<dim3(M / 64, H3 / 64, 1), 256, 0, stream>>>(
        E, EP, 0, wihT[i], H3, 0, bih[i], xp[i], 0, M, H3, EP, M, 0, H3);
  }

  // 4) GRU scans (4 concurrent workgroups, 512 threads each)
  ScanArgs sa;
  for (int i = 0; i < 4; ++i) { sa.xp[i] = xp[i]; sa.whhT[i] = whhT[i]; sa.bhh[i] = bhh[i]; }
  sa.cat[0] = catT; sa.cat[1] = catT; sa.cat[2] = catN; sa.cat[3] = catN;
  sa.hid[0] = hidT; sa.hid[1] = hidT + (size_t)BB * HH;
  sa.hid[2] = hidN; sa.hid[3] = hidN + (size_t)BB * HH;
  sa.len[0] = len_t; sa.len[1] = len_t; sa.len[2] = len_n; sa.len[3] = len_n;
  sa.T[0] = STT; sa.T[1] = STT; sa.T[2] = SNN; sa.T[3] = SNN;
  sa.rev[0] = 0; sa.rev[1] = 1; sa.rev[2] = 0; sa.rev[3] = 1;
  sa.coloff[0] = 0; sa.coloff[1] = HH; sa.coloff[2] = 0; sa.coloff[3] = HH;
  gru_scan<<<4, 512, 0, stream>>>(sa);

  // 5) biatt projections, stored transposed f16 per batch: tnT[b][d][n], ttT[b][d][t]
  gemm_wmma<EP_F16T><<<dim3(MN / 64, H2 / 64, 1), 256, 0, stream>>>(
      catN, H4, 0, biattT, H2, 0, biatt_b, tnT, 0, MN, H2, H2,
      SNN, (long)H2 * SNN, SNN);
  gemm_wmma<EP_F16T><<<dim3(MT / 64, H2 / 64, 1), 256, 0, stream>>>(
      catT, H4, 0, biattT, H2, 0, biatt_b, ttT, 0, MT, H2, H2,
      STT, (long)H2 * STT, STT);

  // 6) batched attention scores
  gemm_wmma<EP_F32><<<dim3(STT / 64, SNN / 64, BB), 256, 0, stream>>>(
      catT, H4, (long)STT * H4, tnT, SNN, (long)H2 * SNN, nullptr,
      scTN, (long)STT * SNN, STT, SNN, H2, STT, 0, SNN);
  gemm_wmma<EP_F32><<<dim3(SNN / 64, STT / 64, BB), 256, 0, stream>>>(
      catN, H4, (long)SNN * H4, ttT, STT, (long)H2 * STT, nullptr,
      scNT, (long)SNN * STT, SNN, STT, H2, SNN, 0, STT);

  // 7) softmax -> f16 probabilities
  softmax_rows<<<BB * STT, 256, 0, stream>>>(scTN, pTN, SNN);
  softmax_rows<<<BB * SNN, 256, 0, stream>>>(scNT, pNT, STT);

  // 8) contexts, written into second half of concat buffers (row stride 2048)
  gemm_wmma<EP_F16><<<dim3(STT / 64, H2 / 64, BB), 256, 0, stream>>>(
      pTN, SNN, (long)STT * SNN, catN, H4, (long)SNN * H4, nullptr,
      catT + H2, (long)STT * H4, STT, H2, SNN, STT, 0, H4);
  gemm_wmma<EP_F16><<<dim3(SNN / 64, H2 / 64, BB), 256, 0, stream>>>(
      pNT, STT, (long)SNN * STT, catT, H4, (long)STT * H4, nullptr,
      catN + H2, (long)SNN * H4, SNN, H2, STT, SNN, 0, H4);

  // 9) v_t / v_n = tanh(cat @ W^T + b) -> directly into d_out [B,640,1024]
  float* out_f = (float*)d_out;
  gemm_wmma<EP_TANH_F32><<<dim3(MT / 64, H2 / 64, 1), 256, 0, stream>>>(
      catT, H4, 0, wtT, H2, 0, wt_b, out_f, 0, MT, H2, H4,
      STT, (long)OUTROW * H2, H2);
  gemm_wmma<EP_TANH_F32><<<dim3(MN / 64, H2 / 64, 1), 256, 0, stream>>>(
      catN, H4, 0, wnT, H2, 0, wn_b, out_f + (size_t)STT * H2, 0, MN, H2, H4,
      SNN, (long)OUTROW * H2, H2);

  // 10) hidden = comb(cat([hid_t,hid_n],-1)) -> d_out tail [2,B,512]
  comb_kernel<<<cgrid(2 * BB * HH), 256, 0, stream>>>(
      hidT, hidN, comb_W, comb_b, out_f + (size_t)BB * OUTROW * H2);
}
